// TransformerContinuous_46583215292818
// MI455X (gfx1250) — compile-verified, matrix operand order checked by reading, not run on hardware
//
#include <hip/hip_runtime.h>
#include <cstdint>

typedef __bf16 bf16_t;
typedef __attribute__((ext_vector_type(16))) bf16_t v16bf;
typedef __attribute__((ext_vector_type(8)))  bf16_t v8bf;
typedef __attribute__((ext_vector_type(8)))  float  v8f;
typedef __attribute__((__vector_size__(16))) int    i32x4;

#define AS1 __attribute__((address_space(1)))
#define AS3 __attribute__((address_space(3)))

#if __has_builtin(__builtin_amdgcn_global_load_async_to_lds_b128)
#define HAVE_ASYNC_LDS 1
#else
#define HAVE_ASYNC_LDS 0
#endif

#define NB 8
#define LX 512
#define NWTOK 384
#define SEQ 896
#define DIM 1024
#define NH 16
#define HD 64
#define FFD 4096
#define NLAY 4
#define HAM 144
#define MROWS (NB*SEQ)   // 7168

// ---------- helpers ----------

__device__ __forceinline__ bf16_t to_bf(float f) {
  unsigned u = __float_as_uint(f);
  unsigned r = (u + 0x7fffu + ((u >> 16) & 1u)) >> 16;
  unsigned short s = (unsigned short)r;
  bf16_t b;
  __builtin_memcpy(&b, &s, 2);
  return b;
}

// Build a 16-element bf16 fragment from two 16B chunks
// (CDNA5 16-bit A/B fragment layout: two 8-element K-chunks per lane)
__device__ __forceinline__ v16bf ld_frag(const bf16_t* p0, const bf16_t* p1) {
  v8bf lo = *(const v8bf*)p0;
  v8bf hi = *(const v8bf*)p1;
  return __builtin_shufflevector(lo, hi, 0,1,2,3,4,5,6,7,8,9,10,11,12,13,14,15);
}

__device__ __forceinline__ v8f wmma_bf16(v16bf a, v16bf b, v8f c) {
  return __builtin_amdgcn_wmma_f32_16x16x32_bf16(false, a, false, b, (short)0, c, false, false);
}

template<int N>
__device__ __forceinline__ void wait_async() {
#if __has_builtin(__builtin_amdgcn_s_wait_asynccnt)
  __builtin_amdgcn_s_wait_asynccnt(N);
#elif HAVE_ASYNC_LDS
  asm volatile("s_wait_asynccnt %0" :: "i"(N) : "memory");
#endif
}

__device__ __forceinline__ void async_b128(const void* g, void* l) {
#if HAVE_ASYNC_LDS
  i32x4* gg = (i32x4*)g;   // strip const / set pointee type (generic AS)
  i32x4* ll = (i32x4*)l;
  __builtin_amdgcn_global_load_async_to_lds_b128((AS1 i32x4*)gg, (AS3 i32x4*)ll, 0, 0);
#else
  (void)g; (void)l;
#endif
}

// ---------- fp32 -> bf16 weight conversion ----------

__global__ __launch_bounds__(256) void cvt_bf16_kernel(const float* __restrict__ in,
                                                       bf16_t* __restrict__ out, long n) {
  long i = (long)blockIdx.x * 256 + threadIdx.x;
  long stride = (long)gridDim.x * 256;
  for (; i < n; i += stride) out[i] = to_bf(in[i]);
}

// ---------- main GEMM: C[M,N] = A[M,K] @ W[N,K]^T + bias, optional exact GELU ----------
// 128x128 block tile, BK=64, double-buffered LDS fed by async-to-LDS loads.
// 8 wave32 waves: each wave 64x32 = 4x2 WMMA tiles; 16 WMMAs per K iteration.
// All 12 fragment loads of a k-step are hoisted ahead of the 8 WMMAs so the
// ds_loads pipeline instead of serializing on s_wait_dscnt per WMMA pair.

__global__ __launch_bounds__(256) void gemm_bf16_kernel(
    const bf16_t* __restrict__ A, const bf16_t* __restrict__ W,
    const float* __restrict__ bias, float* __restrict__ Cf,
    bf16_t* __restrict__ Cb, int M, int N, int K, int do_gelu)
{
  __shared__ __align__(16) bf16_t sA[2][128][72];
  __shared__ __align__(16) bf16_t sB[2][128][72];
  const int tid  = threadIdx.x;
  const int nT   = N >> 7;
  const int bn   = blockIdx.x % nT;
  const int bm   = blockIdx.x / nT;
  const int wid  = tid >> 5, lane = tid & 31;
  const int wm   = wid >> 2, wn   = wid & 3;
  const int half = lane >> 4, lr  = lane & 15;

  const int srow = tid >> 1;          // 0..127
  const int scol = (tid & 1) * 32;    // 0 or 32
  const bf16_t* Ag = A + (size_t)(bm * 128) * K;
  const bf16_t* Wg = W + (size_t)(bn * 128) * K;

  auto stage = [&](int buf, int kb) {
#if HAVE_ASYNC_LDS
#pragma unroll
    for (int c = 0; c < 4; ++c) {
      async_b128(Ag + (size_t)srow * K + kb + scol + c*8, &sA[buf][srow][scol + c*8]);
      async_b128(Wg + (size_t)srow * K + kb + scol + c*8, &sB[buf][srow][scol + c*8]);
    }
#else
#pragma unroll
    for (int c = 0; c < 4; ++c) {
      *(uint4*)(&sA[buf][srow][scol + c*8]) = *(const uint4*)(Ag + (size_t)srow * K + kb + scol + c*8);
      *(uint4*)(&sB[buf][srow][scol + c*8]) = *(const uint4*)(Wg + (size_t)srow * K + kb + scol + c*8);
    }
#endif
  };

  v8f acc[4][2] = {};

  stage(0, 0);
  int buf = 0;
  for (int kb = 0; kb < K; kb += 64) {
    const bool hasNext = (kb + 64 < K);
    if (hasNext) stage(buf ^ 1, kb + 64);
#if HAVE_ASYNC_LDS
    if (hasNext) wait_async<8>(); else wait_async<0>();
#endif
    __syncthreads();
#pragma unroll
    for (int s = 0; s < 2; ++s) {
      v16bf bfr[2], afr[4];
#pragma unroll
      for (int nt = 0; nt < 2; ++nt) {
        const bf16_t* p = &sB[buf][wn*32 + nt*16 + lr][32*s];
        bfr[nt] = ld_frag(p + 8*half, p + 16 + 8*half);
      }
#pragma unroll
      for (int mt = 0; mt < 4; ++mt) {
        const bf16_t* p = &sA[buf][wm*64 + mt*16 + lr][32*s];
        afr[mt] = ld_frag(p + 8*half, p + 16 + 8*half);
      }
#pragma unroll
      for (int mt = 0; mt < 4; ++mt)
#pragma unroll
        for (int nt = 0; nt < 2; ++nt)
          acc[mt][nt] = wmma_bf16(afr[mt], bfr[nt], acc[mt][nt]);
    }
    __syncthreads();
    buf ^= 1;
  }

#pragma unroll
  for (int mt = 0; mt < 4; ++mt) {
#pragma unroll
    for (int nt = 0; nt < 2; ++nt) {
      int col = bn*128 + wn*32 + nt*16 + lr;
      float bb = bias[col];
#pragma unroll
      for (int g = 0; g < 8; ++g) {
        int row = bm*128 + wm*64 + mt*16 + g + 8*half;
        float v = acc[mt][nt][g] + bb;
        if (do_gelu) v = 0.5f * v * (1.0f + erff(v * 0.70710678118f));
        size_t idx = (size_t)row * N + col;
        if (Cf) Cf[idx] = v;
        if (Cb) Cb[idx] = to_bf(v);
      }
    }
  }
}

// ---------- flash attention (WMMA, online softmax) ----------
// Block = 128 queries of one (b,h); 8 waves x 16 queries. K/V streamed in 32-key chunks.

__global__ __launch_bounds__(256) void flash_attn_kernel(
    const bf16_t* __restrict__ qkv, const float* __restrict__ xmask,
    bf16_t* __restrict__ obf)
{
  __shared__ __align__(16) bf16_t sK[32][64];     // [key][d]
  __shared__ __align__(16) bf16_t sVt[64][32];    // [d][key] (transposed)
  __shared__ __align__(16) bf16_t sP[8][16][32];  // per-wave P tile

  const int qTiles = SEQ / 128;   // 7
  int qb = blockIdx.x % qTiles;
  int bh = blockIdx.x / qTiles;
  int b = bh / NH, h = bh % NH;

  const int tid  = threadIdx.x;
  const int wid  = tid >> 5, lane = tid & 31;
  const int half = lane >> 4, lr  = lane & 15;
  const int q0   = qb*128 + wid*16;

  // Q fragments (K-dim = 64 -> 2 WMMA k-steps), loaded directly from global
  const bf16_t* qrow = qkv + ((size_t)(b*SEQ + q0 + lr) * (3*DIM)) + h*HD;
  v16bf aQ[2];
#pragma unroll
  for (int s = 0; s < 2; ++s)
    aQ[s] = ld_frag(qrow + 32*s + 8*half, qrow + 32*s + 16 + 8*half);

  v8f oacc[4] = {};
  float mrow[8], lsum[8];
#pragma unroll
  for (int g = 0; g < 8; ++g) { mrow[g] = -1e30f; lsum[g] = 0.f; }

  const int krow = tid >> 3;        // 0..31 key
  const int kcol = (tid & 7) * 8;   // 0..56 d

  for (int kb = 0; kb < SEQ; kb += 32) {
    __syncthreads();
    {   // stage K chunk (async direct-to-LDS) and transposed V chunk
      const bf16_t* kp = qkv + ((size_t)(b*SEQ + kb + krow) * (3*DIM)) + DIM   + h*HD + kcol;
      const bf16_t* vp = qkv + ((size_t)(b*SEQ + kb + krow) * (3*DIM)) + 2*DIM + h*HD + kcol;
#if HAVE_ASYNC_LDS
      async_b128(kp, &sK[krow][kcol]);
#else
      *(uint4*)(&sK[krow][kcol]) = *(const uint4*)kp;
#endif
      v8bf vv = *(const v8bf*)vp;
#pragma unroll
      for (int j = 0; j < 8; ++j) sVt[kcol + j][krow] = vv[j];
#if HAVE_ASYNC_LDS
      wait_async<0>();
#endif
    }
    __syncthreads();

    bool active = (kb < LX) || (kb <= q0 + 15);
    if (active) {
      // hoist all 4 K fragments, then run the 4 score WMMAs
      v16bf bK[2][2];
#pragma unroll
      for (int t = 0; t < 2; ++t)
#pragma unroll
        for (int s = 0; s < 2; ++s) {
          const bf16_t* p = &sK[t*16 + lr][32*s];
          bK[t][s] = ld_frag(p + 8*half, p + 16 + 8*half);
        }
      float psc[2][8];
#pragma unroll
      for (int t = 0; t < 2; ++t) {
        v8f sc = {};
#pragma unroll
        for (int s = 0; s < 2; ++s) sc = wmma_bf16(aQ[s], bK[t][s], sc);
        int col = kb + t*16 + lr;
        bool colmask = (col < LX) ? (xmask[b*LX + col] != 0.0f) : false;
#pragma unroll
        for (int g = 0; g < 8; ++g) {
          int row = q0 + g + 8*half;
          float v = sc[g] * 0.125f;   // 1/sqrt(64)
          if (colmask || (col >= LX && col > row)) v = -1e30f;
          psc[t][g] = v;
        }
      }
      // online softmax over the 32 keys (columns live across a 16-lane group)
#pragma unroll
      for (int g = 0; g < 8; ++g) {
        float mx = fmaxf(psc[0][g], psc[1][g]);
#pragma unroll
        for (int d = 1; d < 16; d <<= 1) mx = fmaxf(mx, __shfl_xor(mx, d, 32));
        float mnew  = fmaxf(mrow[g], mx);
        float alpha = __expf(mrow[g] - mnew);
        float p0 = __expf(psc[0][g] - mnew);
        float p1 = __expf(psc[1][g] - mnew);
        float ps = p0 + p1;
#pragma unroll
        for (int d = 1; d < 16; d <<= 1) ps += __shfl_xor(ps, d, 32);
        lsum[g] = lsum[g] * alpha + ps;
        mrow[g] = mnew;
#pragma unroll
        for (int nt = 0; nt < 4; ++nt) oacc[nt][g] *= alpha;
        int rloc = g + 8*half;
        sP[wid][rloc][lr]      = to_bf(p0);
        sP[wid][rloc][16 + lr] = to_bf(p1);
      }
    }
    __syncthreads();   // uniform barrier; makes P visible for A-fragment reload
    if (active) {
      const bf16_t* pp = &sP[wid][lr][0];
      v16bf aP = ld_frag(pp + 8*half, pp + 16 + 8*half);
      v16bf bV[4];
#pragma unroll
      for (int nt = 0; nt < 4; ++nt) {
        const bf16_t* vp2 = &sVt[nt*16 + lr][0];
        bV[nt] = ld_frag(vp2 + 8*half, vp2 + 16 + 8*half);
      }
#pragma unroll
      for (int nt = 0; nt < 4; ++nt)
        oacc[nt] = wmma_bf16(aP, bV[nt], oacc[nt]);
    }
  }

#pragma unroll
  for (int g = 0; g < 8; ++g) {
    int row = q0 + g + 8*half;
    float inv = 1.0f / (lsum[g] + 1e-30f);
#pragma unroll
    for (int nt = 0; nt < 4; ++nt) {
      int col = h*HD + nt*16 + lr;
      obf[(size_t)(b*SEQ + row) * DIM + col] = to_bf(oacc[nt][g] * inv);
    }
  }
}

// ---------- residual + LayerNorm (row-wise, D=1024, 4 elems/thread) ----------

__global__ __launch_bounds__(256) void add_ln_kernel(
    float* __restrict__ tgt, const float* __restrict__ delta,
    const float* __restrict__ g, const float* __restrict__ be,
    bf16_t* __restrict__ tgt_bf)
{
  __shared__ float sred[256];
  const int row = blockIdx.x;
  const int tid = threadIdx.x;
  float v[4];
#pragma unroll
  for (int i = 0; i < 4; ++i) {
    int c = tid + i*256;
    v[i] = tgt[(size_t)row*DIM + c] + delta[(size_t)row*DIM + c];
  }
  float s = v[0]+v[1]+v[2]+v[3];
  sred[tid] = s; __syncthreads();
  for (int st = 128; st > 0; st >>= 1) { if (tid < st) sred[tid] += sred[tid+st]; __syncthreads(); }
  float mean = sred[0] / (float)DIM; __syncthreads();
  float sq = 0.f;
#pragma unroll
  for (int i = 0; i < 4; ++i) { float d = v[i]-mean; sq += d*d; }
  sred[tid] = sq; __syncthreads();
  for (int st = 128; st > 0; st >>= 1) { if (tid < st) sred[tid] += sred[tid+st]; __syncthreads(); }
  float rstd = rsqrtf(sred[0] / (float)DIM + 1e-5f);
#pragma unroll
  for (int i = 0; i < 4; ++i) {
    int c = tid + i*256;
    float o = (v[i]-mean)*rstd*g[c] + be[c];
    size_t idx = (size_t)row*DIM + c;
    tgt[idx] = o; tgt_bf[idx] = to_bf(o);
  }
}

// ---------- embeddings ----------

__global__ __launch_bounds__(256) void embed_x_kernel(
    const float* __restrict__ x, const float* __restrict__ Wn,
    const float* __restrict__ bn, float* __restrict__ tgt,
    bf16_t* __restrict__ tgt_bf, float* __restrict__ xmask)
{
  __shared__ float sa[63];
  __shared__ float st;
  int r = blockIdx.x;                  // 0..B*LX-1
  int b = r / LX, i = r % LX;
  int tid = threadIdx.x;
  if (tid < 64) {
    float xv = x[(size_t)r * 64 + tid];
    float xc = isnan(xv) ? 0.f : xv;
    float nrm = xc * 0.99995000375f;   // 1/sqrt(1+EPS_RMS)
    nrm = fminf(5.f, fmaxf(-5.f, nrm));
    if (tid == 0) { st = xc; xmask[r] = isnan(xv) ? 1.f : 0.f; }
    else sa[tid-1] = nrm;
  }
  __syncthreads();
  float t = st;
#pragma unroll
  for (int u = 0; u < 4; ++u) {
    int c = tid * 4 + u;
    const float* wr = Wn + (size_t)c * 63;
    float acc = bn[c];
    for (int k = 0; k < 63; ++k) acc += sa[k] * wr[k];
    int j = c >> 1;
    float ang = t * __expf(-0.00674585477f * (float)j);  // ln(1000)/1024
    float pe = (c & 1) ? __cosf(ang) : __sinf(ang);
    float o = acc + pe;
    size_t idx = (size_t)(b*SEQ + i) * DIM + c;
    tgt[idx] = o; tgt_bf[idx] = to_bf(o);
  }
}

__global__ __launch_bounds__(256) void embed_w_kernel(
    const float* __restrict__ w, const float* __restrict__ Wc,
    const float* __restrict__ bc, float* __restrict__ tgt,
    bf16_t* __restrict__ tgt_bf)
{
  __shared__ float sa[6];
  int r = blockIdx.x;                  // 0..B*384-1
  int b = r / NWTOK, j = r % NWTOK;
  int tid = threadIdx.x;
  if (tid < 6) {
    float wv = w[(size_t)r * 6 + tid];
    float wc = isnan(wv) ? 0.f : wv;
    sa[tid] = fminf(5.f, fmaxf(-5.f, wc * 0.99995000375f));
  }
  __syncthreads();
  float pos = (float)(LX + j);
#pragma unroll
  for (int u = 0; u < 4; ++u) {
    int c = tid * 4 + u;
    const float* wr = Wc + (size_t)c * 6;
    float acc = bc[c];
#pragma unroll
    for (int k = 0; k < 6; ++k) acc += sa[k]*wr[k];
    int jj = c >> 1;
    float ang = pos * __expf(-0.00674585477f * (float)jj);
    float pe = (c & 1) ? __cosf(ang) : __sinf(ang);
    float o = acc + pe;
    size_t idx = (size_t)(b*SEQ + LX + j) * DIM + c;
    tgt[idx] = o; tgt_bf[idx] = to_bf(o);
  }
}

// ---------- head + loss ----------

__global__ __launch_bounds__(256) void head_loss_kernel(
    const float* __restrict__ tgt, const float* __restrict__ Wh,
    const float* __restrict__ bh, const float* __restrict__ y,
    const float* __restrict__ w, float* __restrict__ partial)
{
  __shared__ float sred[256];
  int b = blockIdx.x, tid = threadIdx.x;
  float sq = 0.f;
  if (tid < HAM) {
    const float* tr = tgt + (size_t)(b*SEQ + SEQ-1) * DIM;
    const float* wr = Wh + (size_t)tid * DIM;
    float acc = bh[tid];
    for (int k = 0; k < DIM; ++k) acc += tr[k]*wr[k];
    float resid = y[b*HAM + tid] - w[((size_t)b*16 + 15)*HAM + tid];  // w_last
    float e = acc - resid;
    sq = e*e;
  }
  sred[tid] = sq; __syncthreads();
  for (int st = 128; st > 0; st >>= 1) { if (tid < st) sred[tid] += sred[tid+st]; __syncthreads(); }
  if (tid == 0) partial[b] = sred[0];
}

__global__ void finalize_kernel(const float* __restrict__ partial, float* __restrict__ out) {
  if (threadIdx.x == 0) {
    float s = 0.f;
    for (int i = 0; i < NB; ++i) s += partial[i];
    out[0] = s / (float)(NB * HAM);
  }
}

// ---------- host ----------

extern "C" void kernel_launch(void* const* d_in, const int* in_sizes, int n_in,
                              void* d_out, int out_size, void* d_ws, size_t ws_size,
                              hipStream_t stream)
{
  (void)in_sizes; (void)n_in; (void)out_size; (void)ws_size;
  const float* x      = (const float*)d_in[0];
  const float* w      = (const float*)d_in[1];
  const float* y      = (const float*)d_in[2];
  const float* W_nail = (const float*)d_in[3];
  const float* b_nail = (const float*)d_in[4];
  const float* W_cond = (const float*)d_in[5];
  const float* b_cond = (const float*)d_in[6];
  const float* Wqkv   = (const float*)d_in[7];
  const float* bqkv   = (const float*)d_in[8];
  const float* Wo     = (const float*)d_in[9];
  const float* bo     = (const float*)d_in[10];
  const float* ln1_g  = (const float*)d_in[11];
  const float* ln1_b  = (const float*)d_in[12];
  const float* ln2_g  = (const float*)d_in[13];
  const float* ln2_b  = (const float*)d_in[14];
  const float* W1     = (const float*)d_in[15];
  const float* b1     = (const float*)d_in[16];
  const float* W2     = (const float*)d_in[17];
  const float* b2     = (const float*)d_in[18];
  const float* W_ham  = (const float*)d_in[19];
  const float* b_ham  = (const float*)d_in[20];

  char* ws = (char*)d_ws;
  size_t off = 0;
  auto alloc = [&](size_t bytes) -> void* {
    void* p = ws + off;
    off = (off + bytes + 255) & ~(size_t)255;
    return p;
  };

  const size_t nWq = (size_t)NLAY*3*DIM*DIM;
  const size_t nWo = (size_t)NLAY*DIM*DIM;
  const size_t nW1 = (size_t)NLAY*FFD*DIM;
  const size_t nW2 = (size_t)NLAY*DIM*FFD;

  bf16_t* wq_bf  = (bf16_t*)alloc(nWq*2);
  bf16_t* wo_bf  = (bf16_t*)alloc(nWo*2);
  bf16_t* w1_bf  = (bf16_t*)alloc(nW1*2);
  bf16_t* w2_bf  = (bf16_t*)alloc(nW2*2);
  float*  tgt    = (float*)alloc((size_t)MROWS*DIM*4);
  bf16_t* tgt_bf = (bf16_t*)alloc((size_t)MROWS*DIM*2);
  bf16_t* qkv_bf = (bf16_t*)alloc((size_t)MROWS*3*DIM*2);
  bf16_t* o_bf   = (bf16_t*)alloc((size_t)MROWS*DIM*2);
  bf16_t* f1_bf  = (bf16_t*)alloc((size_t)MROWS*FFD*2);
  float*  tmp    = (float*)alloc((size_t)MROWS*DIM*4);
  float*  xmask  = (float*)alloc((size_t)NB*LX*4);
  float*  part   = (float*)alloc(64*4);

  cvt_bf16_kernel<<<4096, 256, 0, stream>>>(Wqkv, wq_bf, (long)nWq);
  cvt_bf16_kernel<<<2048, 256, 0, stream>>>(Wo,   wo_bf, (long)nWo);
  cvt_bf16_kernel<<<4096, 256, 0, stream>>>(W1,   w1_bf, (long)nW1);
  cvt_bf16_kernel<<<4096, 256, 0, stream>>>(W2,   w2_bf, (long)nW2);

  embed_x_kernel<<<NB*LX,    256, 0, stream>>>(x, W_nail, b_nail, tgt, tgt_bf, xmask);
  embed_w_kernel<<<NB*NWTOK, 256, 0, stream>>>(w, W_cond, b_cond, tgt, tgt_bf);

  const int gQKV = (MROWS/128) * (3*DIM/128);
  const int gWO  = (MROWS/128) * (DIM/128);
  const int gF1  = (MROWS/128) * (FFD/128);
  const int gAtt = (SEQ/128) * NB * NH;

  for (int l = 0; l < NLAY; ++l) {
    gemm_bf16_kernel<<<gQKV, 256, 0, stream>>>(tgt_bf, wq_bf + (size_t)l*3*DIM*DIM,
        bqkv + (size_t)l*3*DIM, nullptr, qkv_bf, MROWS, 3*DIM, DIM, 0);
    flash_attn_kernel<<<gAtt, 256, 0, stream>>>(qkv_bf, xmask, o_bf);
    gemm_bf16_kernel<<<gWO, 256, 0, stream>>>(o_bf, wo_bf + (size_t)l*DIM*DIM,
        bo + (size_t)l*DIM, tmp, nullptr, MROWS, DIM, DIM, 0);
    add_ln_kernel<<<MROWS, 256, 0, stream>>>(tgt, tmp, ln1_g + l*DIM, ln1_b + l*DIM, tgt_bf);
    gemm_bf16_kernel<<<gF1, 256, 0, stream>>>(tgt_bf, w1_bf + (size_t)l*FFD*DIM,
        b1 + (size_t)l*FFD, nullptr, f1_bf, MROWS, FFD, DIM, 1);
    gemm_bf16_kernel<<<gWO, 256, 0, stream>>>(f1_bf, w2_bf + (size_t)l*DIM*FFD,
        b2 + (size_t)l*DIM, tmp, nullptr, MROWS, DIM, FFD, 0);
    add_ln_kernel<<<MROWS, 256, 0, stream>>>(tgt, tmp, ln2_g + l*DIM, ln2_b + l*DIM, tgt_bf);
  }

  head_loss_kernel<<<NB, 256, 0, stream>>>(tgt, W_ham, b_ham, y, w, part);
  finalize_kernel<<<1, 32, 0, stream>>>(part, (float*)d_out);
}
